// FusedComplexLinearAttention_28286654611777
// MI455X (gfx1250) — matrix operand order repaired and checked by reading, not compiled, not run on hardware
//
#include <hip/hip_runtime.h>

// ---------------------------------------------------------------------------
// Problem constants (match reference)
// ---------------------------------------------------------------------------
#define B_   2
#define S_   4096
#define HID_ 4096
#define H_   32
#define D_   128
#define dH_  64          // complex half dim
#define C_   256         // chunk size
#define N_   16          // S_/C_
#define BH_  64          // B_*H_
#define BS_  8192        // B_*S_

typedef __attribute__((ext_vector_type(8)))  __bf16 v8bf;
typedef __attribute__((ext_vector_type(16))) __bf16 v16bf;
typedef __attribute__((ext_vector_type(8)))  float  v8f;
typedef __attribute__((ext_vector_type(4)))  unsigned int v4u;
typedef __attribute__((ext_vector_type(8)))  int v8i;
typedef __attribute__((ext_vector_type(4)))  int v4i;

static __device__ inline v8f zero8() {
    v8f z;
#pragma unroll
    for (int i = 0; i < 8; ++i) z[i] = 0.0f;
    return z;
}

static __device__ inline v16bf mk16(v8bf a, v8bf b) {
    v16bf r;
#pragma unroll
    for (int i = 0; i < 8; ++i) { r[i] = a[i]; r[i + 8] = b[i]; }
    return r;
}

// D = A(16x32 bf16) * B(32x16 bf16) + C (f32)
static __device__ inline v8f wmma_bf16(v16bf a, v16bf b, v8f c) {
    return __builtin_amdgcn_wmma_f32_16x16x32_bf16(
        false, a, false, b, (short)0, c, false, false);
}

// A-matrix fragment from row-major M[row][k], leading dim ld (elements).
// Lane layout (ISA 7.12.2, 16-bit A 16x32): lanes 0-15 hold K {0..7,16..23},
// lanes 16-31 hold K {8..15,24..31} of row (lane&15).
static __device__ inline v16bf frag_a_g(const __bf16* M, int ld, int row0,
                                        int k0, int lane) {
    int lo = lane & 15, hi = lane >> 4;
    const __bf16* p = M + (size_t)(row0 + lo) * ld + k0 + hi * 8;
    return mk16(*(const v8bf*)p, *(const v8bf*)(p + 16));
}

// B-matrix fragment where B[k][n] = M[n][k] (M row-major, leading dim ld).
// Lane layout: lanes 0-15 hold K {0..7,16..23}, lanes 16-31 hold K {8..15,24..31}, col (lane&15).
static __device__ inline v16bf frag_b_g(const __bf16* M, int ld, int col0,
                                        int k0, int lane) {
    int lo = lane & 15, hi = lane >> 4;
    const __bf16* p = M + (size_t)(col0 + lo) * ld + k0 + hi * 8;
    return mk16(*(const v8bf*)p, *(const v8bf*)(p + 16));
}

// ---------------------------------------------------------------------------
// TDM: issue a 2-D bf16 tile load (128 rows x 64 cols, row stride HID_) into
// LDS. Descriptor per CDNA5 ISA ch.8 (D# group0/group1 bit layout). LDS rows
// are padded by the TDM itself: pad_interval = 32 DWORDs (= one 64-elem bf16
// row), pad_amount = 4 DWORDs -> LDS row stride 144B = 72 elements.
// This toolchain exposes the 6-arg builtin:
//   (v4u g0, v8i g1, v4i g2, v4i g3, v8i extra, i32 cpol)
// ---------------------------------------------------------------------------
#define KT_   64                 // K-tile (elements)
#define LDT_  72                 // padded LDS row stride (elements)

static __device__ inline void tdm_load_tile(const __bf16* gsrc,
                                            unsigned lds_byte_off) {
    unsigned long long ga = (unsigned long long)(uintptr_t)gsrc;
    v4u g0 = { 1u,                                   // count=1, user desc
               lds_byte_off,                         // lds_addr (bytes)
               (unsigned)(ga & 0xffffffffu),         // global_addr[31:0]
               (unsigned)(ga >> 32) | (2u << 30) };  // global_addr[56:32]|type=2
    v8i g1;
    g1[0] = (1 << 16)            // data_size = 1 -> 2 bytes
          | (1 << 20)            // pad_enable
          | (4 << 22)            // pad_interval code 4 -> 32 DWORDs
          | (3 << 25);           // pad_amount  code 3 -> 4 DWORDs (16B)
    g1[1] = (int)((HID_ & 0xffff) << 16);            // tensor_dim0 lo16
    g1[2] = (int)((HID_ >> 16) | ((BS_ & 0xffff) << 16)); // td0 hi | td1 lo
    g1[3] = (int)((BS_ >> 16) | (KT_ << 16));        // td1 hi | tile_dim0
    g1[4] = 128;                                     // tile_dim1=128, tile_dim2=0
    g1[5] = HID_;                                    // tensor_dim0_stride lo32
    g1[6] = 0;                                       // stride hi | dim1_stride lo
    g1[7] = 0;
    v4i z4 = { 0, 0, 0, 0 };
    v8i z8 = { 0, 0, 0, 0, 0, 0, 0, 0 };
    __builtin_amdgcn_tensor_load_to_lds(g0, g1, z4, z4, z8, 0);
}

// ---------------------------------------------------------------------------
// K0: fp32 -> bf16 conversion
// ---------------------------------------------------------------------------
__global__ void __launch_bounds__(256)
cvt_f32_bf16(const float* __restrict__ src, __bf16* __restrict__ dst, int n) {
    size_t i = ((size_t)blockIdx.x * 256 + threadIdx.x) * 4;
    if ((int)i + 3 < n) {
        float4 v = *(const float4*)(src + i);
        dst[i + 0] = (__bf16)v.x;
        dst[i + 1] = (__bf16)v.y;
        dst[i + 2] = (__bf16)v.z;
        dst[i + 3] = (__bf16)v.w;
    }
}

// ---------------------------------------------------------------------------
// K1 / K5: out[m,n] = sum_k A[m,k] * W[n,k]   (A: [8192,4096], W: [4096,4096])
// Block tile 128x128, K-tile 64, 256 threads = 8 waves, wave owns 16 columns.
// Tiles are streamed global->LDS by the Tensor Data Mover, double buffered;
// wave 0 issues the DMA and synchronizes with s_wait_tensorcnt.
// ---------------------------------------------------------------------------
template <bool F32OUT>
__global__ void __launch_bounds__(256)
gemm_nt_kernel(const __bf16* __restrict__ A, const __bf16* __restrict__ W,
               __bf16* __restrict__ obf, float* __restrict__ of) {
    __shared__ __align__(16) __bf16 sA[2][128 * LDT_];
    __shared__ __align__(16) __bf16 sB[2][128 * LDT_];

    const int m0 = blockIdx.x * 128, n0 = blockIdx.y * 128;
    const int t = threadIdx.x, lane = t & 31, w = t >> 5;
    const int lo = lane & 15, hi = lane >> 4;

    const unsigned ldsA[2] = { (unsigned)(uintptr_t)&sA[0][0],
                               (unsigned)(uintptr_t)&sA[1][0] };
    const unsigned ldsB[2] = { (unsigned)(uintptr_t)&sB[0][0],
                               (unsigned)(uintptr_t)&sB[1][0] };

    v8f acc[8];
#pragma unroll
    for (int i = 0; i < 8; ++i) acc[i] = zero8();

    constexpr int NKT = HID_ / KT_;   // 64 K-tiles

    if (t < 32) {   // wave 0 primes buffer 0
        tdm_load_tile(A + (size_t)m0 * HID_, ldsA[0]);
        tdm_load_tile(W + (size_t)n0 * HID_, ldsB[0]);
    }

    for (int kt = 0; kt < NKT; ++kt) {
        const int cur = kt & 1;
        if (t < 32) {
            if (kt + 1 < NKT) {   // stream next tile into the other buffer
                tdm_load_tile(A + (size_t)m0 * HID_ + (kt + 1) * KT_,
                              ldsA[cur ^ 1]);
                tdm_load_tile(W + (size_t)n0 * HID_ + (kt + 1) * KT_,
                              ldsB[cur ^ 1]);
                __builtin_amdgcn_s_wait_tensorcnt(2);  // current pair landed
            } else {
                __builtin_amdgcn_s_wait_tensorcnt(0);
            }
        }
        __syncthreads();   // publish LDS tile to all waves

        const __bf16* tb = &sB[cur][0];
        const __bf16* ta = &sA[cur][0];
        v16bf bf[2];
#pragma unroll
        for (int s = 0; s < 2; ++s) {
            const __bf16* p = tb + (size_t)(w * 16 + lo) * LDT_ + s * 32 + hi * 8;
            bf[s] = mk16(*(const v8bf*)p, *(const v8bf*)(p + 16));
        }
#pragma unroll
        for (int mi = 0; mi < 8; ++mi) {
#pragma unroll
            for (int s = 0; s < 2; ++s) {
                const __bf16* p = ta + (size_t)(mi * 16 + lo) * LDT_ + s * 32 + hi * 8;
                acc[mi] = wmma_bf16(mk16(*(const v8bf*)p, *(const v8bf*)(p + 16)),
                                    bf[s], acc[mi]);
            }
        }
        __syncthreads();   // all waves done with `cur` before TDM overwrites it
    }

#pragma unroll
    for (int mi = 0; mi < 8; ++mi) {
#pragma unroll
        for (int r = 0; r < 8; ++r) {
            int grow = m0 + mi * 16 + r + hi * 8;
            int gcol = n0 + w * 16 + lo;
            if (F32OUT) of[(size_t)grow * HID_ + gcol] = acc[mi][r];
            else        obf[(size_t)grow * HID_ + gcol] = (__bf16)acc[mi][r];
        }
    }
}

// ---------------------------------------------------------------------------
// K2: RoPE + split real/imag, build row-major / transposed / negated copies.
// One thread per (b,h,s).
// ---------------------------------------------------------------------------
__global__ void __launch_bounds__(256)
rope_split_kernel(const __bf16* __restrict__ qb, const __bf16* __restrict__ kb,
                  const __bf16* __restrict__ vb, const int* __restrict__ pos,
                  __bf16* __restrict__ qr, __bf16* __restrict__ qi,
                  __bf16* __restrict__ kr, __bf16* __restrict__ ki,
                  __bf16* __restrict__ kin,
                  __bf16* __restrict__ krT, __bf16* __restrict__ kiT,
                  __bf16* __restrict__ vrT, __bf16* __restrict__ viT,
                  __bf16* __restrict__ viTn) {
    int id = blockIdx.x * 256 + threadIdx.x;   // (b*H + h)*S + s
    int s = id & (S_ - 1);
    int bh = id >> 12;
    int h = bh & (H_ - 1), b = bh >> 5;
    int n = s >> 8, cc = s & (C_ - 1);

    float p = (float)pos[b * S_ + s];
    size_t src = ((size_t)b * S_ + s) * HID_ + h * D_;
    size_t drow = (size_t)id * dH_;                           // [BH*S, 64]
    size_t tbase = ((size_t)(bh * N_ + n)) * (dH_ * C_) + cc; // [BH*N][64][256]

    const float LN = 9.210340371976184f; // ln(10000)
#pragma unroll 4
    for (int j = 0; j < dH_; ++j) {
        float invf = expf(-LN * (float)j * (1.0f / 64.0f));
        float f = p * invf, cs, sn;
        __sincosf(f, &sn, &cs);
        float q_im = (float)qb[src + j],  q_re = (float)qb[src + dH_ + j];
        float k_im = (float)kb[src + j],  k_re = (float)kb[src + dH_ + j];
        float v_im = (float)vb[src + j],  v_re = (float)vb[src + dH_ + j];
        float qre2 = q_re * cs + q_im * sn, qim2 = q_im * cs - q_re * sn;
        float kre2 = k_re * cs + k_im * sn, kim2 = k_im * cs - k_re * sn;

        qr[drow + j] = (__bf16)qre2;  qi[drow + j] = (__bf16)qim2;
        kr[drow + j] = (__bf16)kre2;  ki[drow + j] = (__bf16)kim2;
        kin[drow + j] = (__bf16)(-kim2);
        krT [tbase + (size_t)j * C_] = (__bf16)kre2;
        kiT [tbase + (size_t)j * C_] = (__bf16)kim2;
        vrT [tbase + (size_t)j * C_] = (__bf16)v_re;
        viT [tbase + (size_t)j * C_] = (__bf16)v_im;
        viTn[tbase + (size_t)j * C_] = (__bf16)(-v_im);
    }
}

// ---------------------------------------------------------------------------
// K3a: per (b,h,chunk) complex state contribution KV = k^T (x) v  (64x64 x2)
// grid = BH*N blocks, 128 threads = 4 waves, wave owns 16 rows of dd.
// ---------------------------------------------------------------------------
__global__ void __launch_bounds__(128)
chunk_kv_kernel(const __bf16* __restrict__ krT, const __bf16* __restrict__ kiT,
                const __bf16* __restrict__ vrT, const __bf16* __restrict__ viT,
                const __bf16* __restrict__ viTn, float* __restrict__ KV) {
    int blk = blockIdx.x;                       // bh*N + n
    int lane = threadIdx.x & 31, w = threadIdx.x >> 5;
    int lo = lane & 15, hi = lane >> 4;
    const size_t tb = (size_t)blk * (dH_ * C_); // [64][256]

    v8f accR[4], accI[4];
#pragma unroll
    for (int i = 0; i < 4; ++i) { accR[i] = zero8(); accI[i] = zero8(); }

    for (int kk = 0; kk < C_; kk += 32) {
        if (kk + 32 < C_) {   // L2/L0 prefetch of next K-slab fragment rows
            __builtin_prefetch(krT + tb + (size_t)(w * 16 + lo) * C_ + kk + 32, 0, 0);
            __builtin_prefetch(kiT + tb + (size_t)(w * 16 + lo) * C_ + kk + 32, 0, 0);
        }
        v16bf aKr = frag_a_g(krT + tb, C_, w * 16, kk, lane);
        v16bf aKi = frag_a_g(kiT + tb, C_, w * 16, kk, lane);
#pragma unroll
        for (int nt = 0; nt < 4; ++nt) {
            v16bf bVr = frag_b_g(vrT + tb, C_, nt * 16, kk, lane);
            v16bf bVi = frag_b_g(viT + tb, C_, nt * 16, kk, lane);
            v16bf bVn = frag_b_g(viTn + tb, C_, nt * 16, kk, lane);
            accR[nt] = wmma_bf16(aKr, bVr, accR[nt]);
            accR[nt] = wmma_bf16(aKi, bVn, accR[nt]);
            accI[nt] = wmma_bf16(aKr, bVi, accI[nt]);
            accI[nt] = wmma_bf16(aKi, bVr, accI[nt]);
        }
    }
#pragma unroll
    for (int nt = 0; nt < 4; ++nt)
#pragma unroll
        for (int r = 0; r < 8; ++r) {
            int dd = w * 16 + r + hi * 8;
            int e = nt * 16 + lo;
            KV[((size_t)blk * 2 + 0) * 4096 + dd * 64 + e] = accR[nt][r];
            KV[((size_t)blk * 2 + 1) * 4096 + dd * 64 + e] = accI[nt][r];
        }
}

// ---------------------------------------------------------------------------
// K3b: exclusive prefix over chunks; emit bf16 states TRANSPOSED ([e][d]) so
// they load as WMMA B-fragments. grid = BH blocks.
// ---------------------------------------------------------------------------
__global__ void __launch_bounds__(256)
prefix_state_kernel(const float* __restrict__ KV, __bf16* __restrict__ SrT,
                    __bf16* __restrict__ SiT, __bf16* __restrict__ SiTn) {
    int bh = blockIdx.x;
    for (int idx = threadIdx.x; idx < 4096; idx += 256) {
        int dd = idx >> 6, e = idx & 63;
        float aR = 0.0f, aI = 0.0f;
        for (int n = 0; n < N_; ++n) {
            size_t sb = (size_t)(bh * N_ + n) * 4096;
            SrT [sb + e * 64 + dd] = (__bf16)aR;
            SiT [sb + e * 64 + dd] = (__bf16)aI;
            SiTn[sb + e * 64 + dd] = (__bf16)(-aI);
            size_t kb = (size_t)(bh * N_ + n) * 2 * 4096;
            aR += KV[kb + idx];
            aI += KV[kb + 4096 + idx];
        }
    }
}

// ---------------------------------------------------------------------------
// K3c: per (b,h,chunk) output = q@state (inter) + causal intra attention.
// grid = BH*N blocks, 512 threads = 16 waves, wave owns a 16-row M strip.
// Waves diverge on the triangular e-block loop -> NO block barriers; the
// score LDS round-trip is wave-private and ordered with s_wait_dscnt.
// ---------------------------------------------------------------------------
__global__ void __launch_bounds__(512)
chunk_out_kernel(const __bf16* __restrict__ qr, const __bf16* __restrict__ qi,
                 const __bf16* __restrict__ kr, const __bf16* __restrict__ ki,
                 const __bf16* __restrict__ kin,
                 const __bf16* __restrict__ vrT, const __bf16* __restrict__ viT,
                 const __bf16* __restrict__ SrT, const __bf16* __restrict__ SiT,
                 const __bf16* __restrict__ SiTn, float* __restrict__ attn) {
    __shared__ __align__(16) __bf16 sS[16][3][16][32];  // [wave][r/i/-i][row][col]

    int blk = blockIdx.x;                   // bh*N + n
    int bh = blk >> 4, n = blk & 15;
    int b = bh >> 5, h = bh & 31;
    int w = threadIdx.x >> 5, lane = threadIdx.x & 31;
    int lo = lane & 15, hi = lane >> 4;

    // persistent q A-fragments (rows 16w..16w+15 of this chunk)
    const size_t qb = ((size_t)bh * S_ + n * C_ + w * 16) * dH_;
    v16bf aR[2], aI[2];
    aR[0] = frag_a_g(qr + qb, dH_, 0, 0, lane);
    aR[1] = frag_a_g(qr + qb, dH_, 0, 32, lane);
    aI[0] = frag_a_g(qi + qb, dH_, 0, 0, lane);
    aI[1] = frag_a_g(qi + qb, dH_, 0, 32, lane);

    v8f oR[4], oI[4];
#pragma unroll
    for (int i = 0; i < 4; ++i) { oR[i] = zero8(); oI[i] = zero8(); }

    // ---- inter-chunk: o += q @ S_state (complex) ----
    const size_t sb = (size_t)blk * 4096;
#pragma unroll
    for (int kk2 = 0; kk2 < 2; ++kk2) {
#pragma unroll
        for (int nt = 0; nt < 4; ++nt) {
            v16bf bSr = frag_b_g(SrT + sb, 64, nt * 16, kk2 * 32, lane);
            v16bf bSi = frag_b_g(SiT + sb, 64, nt * 16, kk2 * 32, lane);
            v16bf bSn = frag_b_g(SiTn + sb, 64, nt * 16, kk2 * 32, lane);
            oR[nt] = wmma_bf16(aR[kk2], bSr, oR[nt]);
            oR[nt] = wmma_bf16(aI[kk2], bSn, oR[nt]);
            oI[nt] = wmma_bf16(aR[kk2], bSi, oI[nt]);
            oI[nt] = wmma_bf16(aI[kk2], bSr, oI[nt]);
        }
    }

    // ---- intra-chunk, causal, e-blocks of 32 ----
    const size_t kb = ((size_t)bh * S_ + n * C_) * dH_;   // kr/ki rows of chunk
    const size_t vb = (size_t)blk * (dH_ * C_);           // vrT/viT [64][256]
    const int ebmax = (16 * w + 15) >> 5;                 // wave-uniform
    for (int eb = 0; eb <= ebmax; ++eb) {
        if (eb < ebmax) {   // prefetch next e-block's K rows
            __builtin_prefetch(kr + kb + (size_t)(eb * 32 + 32 + lo) * dH_, 0, 0);
            __builtin_prefetch(ki + kb + (size_t)(eb * 32 + 32 + lo) * dH_, 0, 0);
        }
        v8f sRa[2], sIa[2];
#pragma unroll
        for (int i = 0; i < 2; ++i) { sRa[i] = zero8(); sIa[i] = zero8(); }
#pragma unroll
        for (int kk2 = 0; kk2 < 2; ++kk2) {
#pragma unroll
            for (int nt2 = 0; nt2 < 2; ++nt2) {
                int e0 = eb * 32 + nt2 * 16;
                v16bf bKr = frag_b_g(kr + kb, dH_, e0, kk2 * 32, lane);
                v16bf bKi = frag_b_g(ki + kb, dH_, e0, kk2 * 32, lane);
                v16bf bKn = frag_b_g(kin + kb, dH_, e0, kk2 * 32, lane);
                sRa[nt2] = wmma_bf16(aR[kk2], bKr, sRa[nt2]);
                sRa[nt2] = wmma_bf16(aI[kk2], bKn, sRa[nt2]);
                sIa[nt2] = wmma_bf16(aR[kk2], bKi, sIa[nt2]);
                sIa[nt2] = wmma_bf16(aI[kk2], bKr, sIa[nt2]);
            }
        }
        // causal mask + bf16 to wave-private LDS (also store -imag)
#pragma unroll
        for (int nt2 = 0; nt2 < 2; ++nt2)
#pragma unroll
            for (int r = 0; r < 8; ++r) {
                int c = 16 * w + r + hi * 8;
                int e = eb * 32 + nt2 * 16 + lo;
                bool keep = (e <= c);
                float vr_ = keep ? sRa[nt2][r] : 0.0f;
                float vi_ = keep ? sIa[nt2][r] : 0.0f;
                sS[w][0][r + hi * 8][nt2 * 16 + lo] = (__bf16)vr_;
                sS[w][1][r + hi * 8][nt2 * 16 + lo] = (__bf16)vi_;
                sS[w][2][r + hi * 8][nt2 * 16 + lo] = (__bf16)(-vi_);
            }
        asm volatile("s_wait_dscnt 0x0" ::: "memory");

        v16bf aSr = mk16(*(const v8bf*)&sS[w][0][lo][hi * 8],
                         *(const v8bf*)&sS[w][0][lo][hi * 8 + 16]);
        v16bf aSi = mk16(*(const v8bf*)&sS[w][1][lo][hi * 8],
                         *(const v8bf*)&sS[w][1][lo][hi * 8 + 16]);
        v16bf aSn = mk16(*(const v8bf*)&sS[w][2][lo][hi * 8],
                         *(const v8bf*)&sS[w][2][lo][hi * 8 + 16]);
#pragma unroll
        for (int nt = 0; nt < 4; ++nt) {
            v16bf bVr = frag_b_g(vrT + vb, C_, nt * 16, eb * 32, lane);
            v16bf bVi = frag_b_g(viT + vb, C_, nt * 16, eb * 32, lane);
            oR[nt] = wmma_bf16(aSr, bVr, oR[nt]);
            oR[nt] = wmma_bf16(aSn, bVi, oR[nt]);
            oI[nt] = wmma_bf16(aSr, bVi, oI[nt]);
            oI[nt] = wmma_bf16(aSi, bVr, oI[nt]);
        }
    }

    // ---- epilogue: attn[token][h*128 + {real:0..63, imag:64..127}] ----
#pragma unroll
    for (int nt = 0; nt < 4; ++nt)
#pragma unroll
        for (int r = 0; r < 8; ++r) {
            int c = 16 * w + r + hi * 8;
            size_t row = (size_t)b * S_ + n * C_ + c;
            int col = h * D_ + nt * 16 + lo;
            attn[row * HID_ + col]       = oR[nt][r];
            attn[row * HID_ + col + dH_] = oI[nt][r];
        }
}

// ---------------------------------------------------------------------------
// K4: GroupNorm per (token, head) over 128 values; emit bf16 for final GEMM.
// ---------------------------------------------------------------------------
__global__ void __launch_bounds__(256)
groupnorm_kernel(const float* __restrict__ attn, const float* __restrict__ gw,
                 const float* __restrict__ gb, __bf16* __restrict__ out) {
    int tok = blockIdx.x;
    int w = threadIdx.x >> 5, lane = threadIdx.x & 31;
#pragma unroll
    for (int i = 0; i < 4; ++i) {
        int h = w * 4 + i;
        const float* p = attn + (size_t)tok * HID_ + h * D_;
        float x[4], s = 0.0f, ss = 0.0f;
#pragma unroll
        for (int j = 0; j < 4; ++j) {
            x[j] = p[lane * 4 + j];
            s += x[j];
            ss += x[j] * x[j];
        }
#pragma unroll
        for (int off = 16; off > 0; off >>= 1) {
            s += __shfl_xor(s, off);
            ss += __shfl_xor(ss, off);
        }
        float mu = s * (1.0f / 128.0f);
        float var = ss * (1.0f / 128.0f) - mu * mu;
        float rstd = rsqrtf(var + 1e-5f);
#pragma unroll
        for (int j = 0; j < 4; ++j) {
            int col = h * D_ + lane * 4 + j;
            float y = (x[j] - mu) * rstd * gw[col] + gb[col];
            out[(size_t)tok * HID_ + col] = (__bf16)y;
        }
    }
}

// ---------------------------------------------------------------------------
// Host-side launcher
// ---------------------------------------------------------------------------
extern "C" void kernel_launch(void* const* d_in, const int* in_sizes, int n_in,
                              void* d_out, int out_size, void* d_ws, size_t ws_size,
                              hipStream_t stream) {
    const float* hidden = (const float*)d_in[0];
    const int*   pos    = (const int*)d_in[1];
    const float* wq     = (const float*)d_in[2];
    const float* wk     = (const float*)d_in[3];
    const float* wv     = (const float*)d_in[4];
    const float* wo     = (const float*)d_in[5];
    const float* gnw    = (const float*)d_in[6];
    const float* gnb    = (const float*)d_in[7];
    float* out = (float*)d_out;

    size_t off = 0;
    auto alloc = [&](size_t bytes) {
        void* p = (char*)d_ws + off;
        off += (bytes + 255) & ~(size_t)255;
        return p;
    };
    const size_t ACT = (size_t)BS_ * HID_;     // 33.5M elems
    const size_t WEL = (size_t)HID_ * HID_;    // 16.8M elems
    const size_t HSD = (size_t)BH_ * S_ * dH_; // 16.8M elems per half tensor

    __bf16* xbf  = (__bf16*)alloc(ACT * 2);
    __bf16* wqb  = (__bf16*)alloc(WEL * 2);
    __bf16* wkb  = (__bf16*)alloc(WEL * 2);
    __bf16* wvb  = (__bf16*)alloc(WEL * 2);
    __bf16* wob  = (__bf16*)alloc(WEL * 2);
    __bf16* qb   = (__bf16*)alloc(ACT * 2);
    __bf16* kb   = (__bf16*)alloc(ACT * 2);
    __bf16* vb   = (__bf16*)alloc(ACT * 2);
    __bf16* qrp  = (__bf16*)alloc(HSD * 2);
    __bf16* qip  = (__bf16*)alloc(HSD * 2);
    __bf16* krp  = (__bf16*)alloc(HSD * 2);
    __bf16* kip  = (__bf16*)alloc(HSD * 2);
    __bf16* kinp = (__bf16*)alloc(HSD * 2);
    __bf16* krT  = (__bf16*)alloc(HSD * 2);
    __bf16* kiT  = (__bf16*)alloc(HSD * 2);
    __bf16* vrT  = (__bf16*)alloc(HSD * 2);
    __bf16* viT  = (__bf16*)alloc(HSD * 2);
    __bf16* viTn = (__bf16*)alloc(HSD * 2);
    float*  KV   = (float*)alloc((size_t)BH_ * N_ * 2 * 4096 * 4);
    __bf16* SrT  = (__bf16*)alloc((size_t)BH_ * N_ * 4096 * 2);
    __bf16* SiT  = (__bf16*)alloc((size_t)BH_ * N_ * 4096 * 2);
    __bf16* SiTn = (__bf16*)alloc((size_t)BH_ * N_ * 4096 * 2);
    float*  attn = (float*)alloc(ACT * 4);
    __bf16* attb = (__bf16*)alloc(ACT * 2);
    (void)ws_size; (void)n_in; (void)in_sizes; (void)out_size;

    // K0: precision conversion (weights then live bf16 in L2)
    cvt_f32_bf16<<<(int)(ACT / 1024), 256, 0, stream>>>(hidden, xbf, (int)ACT);
    cvt_f32_bf16<<<(int)(WEL / 1024), 256, 0, stream>>>(wq, wqb, (int)WEL);
    cvt_f32_bf16<<<(int)(WEL / 1024), 256, 0, stream>>>(wk, wkb, (int)WEL);
    cvt_f32_bf16<<<(int)(WEL / 1024), 256, 0, stream>>>(wv, wvb, (int)WEL);
    cvt_f32_bf16<<<(int)(WEL / 1024), 256, 0, stream>>>(wo, wob, (int)WEL);

    // K1: QKV projections (TDM-fed WMMA GEMM)
    dim3 gg(BS_ / 128, HID_ / 128);
    gemm_nt_kernel<false><<<gg, 256, 0, stream>>>(xbf, wqb, qb, nullptr);
    gemm_nt_kernel<false><<<gg, 256, 0, stream>>>(xbf, wkb, kb, nullptr);
    gemm_nt_kernel<false><<<gg, 256, 0, stream>>>(xbf, wvb, vb, nullptr);

    // K2: RoPE + operand layout materialization
    rope_split_kernel<<<(B_ * H_ * S_) / 256, 256, 0, stream>>>(
        qb, kb, vb, pos, qrp, qip, krp, kip, kinp, krT, kiT, vrT, viT, viTn);

    // K3: chunk-parallel complex linear attention
    chunk_kv_kernel<<<BH_ * N_, 128, 0, stream>>>(krT, kiT, vrT, viT, viTn, KV);
    prefix_state_kernel<<<BH_, 256, 0, stream>>>(KV, SrT, SiT, SiTn);
    chunk_out_kernel<<<BH_ * N_, 512, 0, stream>>>(
        qrp, qip, krp, kip, kinp, vrT, viT, SrT, SiT, SiTn, attn);

    // K4: GroupNorm per head
    groupnorm_kernel<<<BS_, 256, 0, stream>>>(attn, gnw, gnb, attb);

    // K5: output projection -> fp32 d_out
    gemm_nt_kernel<true><<<gg, 256, 0, stream>>>(attb, wob, nullptr, out);
}
